// MyGNN_70978629533941
// MI455X (gfx1250) — matrix operand order repaired and checked by reading, not compiled, not run on hardware
//
#include <hip/hip_runtime.h>
#include <cstdint>

// ---------------------------------------------------------------------------
// CDNA5 (gfx1250) GINEConv GNN: edge MLPs on v_wmma_f32_16x16x32_bf16,
// fp32 accumulation, fp32 atomic scatter-add.
// ---------------------------------------------------------------------------

typedef __attribute__((ext_vector_type(16))) __bf16 v16bf;
typedef __attribute__((ext_vector_type(8)))  float  v8f;

__device__ __forceinline__ unsigned short f2bf(float f) {
    unsigned int u = __float_as_uint(f);
    u += 0x7FFFu + ((u >> 16) & 1u);          // round-to-nearest-even
    return (unsigned short)(u >> 16);
}

union Frag {
    v16bf          v;
    unsigned int   u32[8];
    unsigned short u16[16];
    uint4          q[2];
};

// ------------------------- small utility kernels ---------------------------

__global__ void zero_f32(float* __restrict__ p, long n) {
    long i = blockIdx.x * (long)blockDim.x + threadIdx.x;
    if (i < n) p[i] = 0.f;
}

__global__ void cvt_f32_bf16(const float* __restrict__ in,
                             unsigned short* __restrict__ out, long n) {
    long i = blockIdx.x * (long)blockDim.x + threadIdx.x;
    if (i < n) out[i] = f2bf(in[i]);
}

// Pack a [K=nkb*32, 128] fp32 weight matrix into bf16 B-fragment order:
// idx = ((t*nkb + kb)*32 + lane)*16 + i  holds  w[K][N],
//   N = t*16 + (lane&15),  K = kb*32 + (i&8 ? 16:0) + (lane&16 ? 8:0) + (i&7)
__global__ void pack_w_bf16(const float* __restrict__ w,
                            unsigned short* __restrict__ out, int nkb) {
    int idx = blockIdx.x * blockDim.x + threadIdx.x;
    if (idx >= nkb * 4096) return;
    int i   = idx & 15;
    int L   = (idx >> 4) & 31;
    int blk = idx >> 9;                 // t*nkb + kb
    int kb  = blk % nkb;
    int t   = blk / nkb;
    int Nn  = t * 16 + (L & 15);
    int K   = kb * 32 + ((i & 8) ? 16 : 0) + ((L & 16) ? 8 : 0) + (i & 7);
    out[idx] = f2bf(w[K * 128 + Nn]);
}

// ------------------------- edge MLP + scatter ------------------------------
// One wave = one 16-edge tile; 4 waves / 128-thread workgroup.

template <int D_NODE>
__global__ __launch_bounds__(128)
void edge_mlp_kernel(const unsigned short* __restrict__ nodef,  // [N][D_NODE] bf16
                     const unsigned short* __restrict__ edgef,  // [E][32]    bf16
                     const int* __restrict__ src,
                     const int* __restrict__ dst,
                     const unsigned short* __restrict__ w1p,    // packed bf16
                     const float* __restrict__ b1,
                     const unsigned short* __restrict__ w2p,    // packed bf16
                     const float* __restrict__ b2,
                     float* __restrict__ accum,                 // [N][128] f32
                     int E)
{
    constexpr int DIN  = D_NODE + 32;
    constexpr int NKB1 = DIN / 32;

    const int wave  = threadIdx.x >> 5;
    const int lane  = threadIdx.x & 31;
    const int g8    = (lane & 16) ? 8 : 0;   // lane-half offset (K and M)
    const int col   = lane & 15;
    const int ebase = (blockIdx.x * 4 + wave) * 16;

    __shared__ __align__(16) unsigned short mt[4][16 * DIN];
    __shared__ __align__(16) unsigned short ht[4][16 * 128];
    unsigned short* mrow = mt[wave];
    unsigned short* hrow = ht[wave];

    // ---- stage gathered [16, DIN] = (node feats || edge feats) into LDS ----
    {
        constexpr int CH = D_NODE / 8;          // uint4 chunks per node row
        for (int c = lane; c < 16 * CH; c += 32) {
            int r = c / CH, k = c - r * CH;
            int e = ebase + r;
            uint4 val = make_uint4(0u, 0u, 0u, 0u);
            if (e < E) {
                long s = src[e];
                val = ((const uint4*)(nodef + s * D_NODE))[k];
            }
            ((uint4*)(mrow + r * DIN))[k] = val;
        }
        for (int c = lane; c < 64; c += 32) {   // 4 chunks per 32-elem edge row
            int r = c >> 2, k = c & 3;
            int e = ebase + r;
            uint4 val = make_uint4(0u, 0u, 0u, 0u);
            if (e < E) val = ((const uint4*)(edgef + (long)e * 32))[k];
            ((uint4*)(mrow + r * DIN + D_NODE))[k] = val;
        }
    }
    asm volatile("" ::: "memory");   // keep DS stores before fragment DS loads

    const v8f vzero = {0.f, 0.f, 0.f, 0.f, 0.f, 0.f, 0.f, 0.f};

    // ---- GEMM1: [16, DIN] x [DIN, 128] -> acc[8] N-tiles ----
    v8f acc[8];
#pragma unroll
    for (int t = 0; t < 8; ++t) acc[t] = vzero;

    const unsigned short* arow = mrow + col * DIN;
#pragma unroll
    for (int kb = 0; kb < NKB1; ++kb) {
        Frag a;
#pragma unroll
        for (int j = 0; j < 8; ++j) {
            int k0 = kb * 32 + ((j & 4) << 2) + g8 + ((j & 3) << 1);
            a.u32[j] = *(const unsigned int*)(arow + k0);
        }
#pragma unroll
        for (int t = 0; t < 8; ++t) {
            Frag b;
            const uint4* bp =
                (const uint4*)(w1p + (((t * NKB1 + kb) * 32 + lane) << 4));
            b.q[0] = bp[0];
            b.q[1] = bp[1];
            acc[t] = __builtin_amdgcn_wmma_f32_16x16x32_bf16(
                false, a.v, false, b.v, (short)0, acc[t], false, false);
        }
    }

    // ---- +b1, ReLU, write hidden tile [16][128] bf16 to LDS ----
#pragma unroll
    for (int t = 0; t < 8; ++t) {
        float bv = b1[t * 16 + col];
#pragma unroll
        for (int r = 0; r < 8; ++r) {
            float v = acc[t][r] + bv;
            v = v > 0.f ? v : 0.f;
            hrow[(r + g8) * 128 + t * 16 + col] = f2bf(v);
        }
    }
    asm volatile("" ::: "memory");

    // ---- GEMM2: [16,128] x [128,128] ----
    v8f acc2[8];
#pragma unroll
    for (int t = 0; t < 8; ++t) acc2[t] = vzero;

    const unsigned short* hr = hrow + col * 128;
#pragma unroll
    for (int kb = 0; kb < 4; ++kb) {
        Frag a;
#pragma unroll
        for (int j = 0; j < 8; ++j) {
            int k0 = kb * 32 + ((j & 4) << 2) + g8 + ((j & 3) << 1);
            a.u32[j] = *(const unsigned int*)(hr + k0);
        }
#pragma unroll
        for (int t = 0; t < 8; ++t) {
            Frag b;
            const uint4* bp =
                (const uint4*)(w2p + (((t * 4 + kb) * 32 + lane) << 4));
            b.q[0] = bp[0];
            b.q[1] = bp[1];
            acc2[t] = __builtin_amdgcn_wmma_f32_16x16x32_bf16(
                false, a.v, false, b.v, (short)0, acc2[t], false, false);
        }
    }

    // ---- +b2, scatter-add into accum[dst] (C/D layout: M = r + g8) ----
    int dv[8];
#pragma unroll
    for (int r = 0; r < 8; ++r) {
        int e = ebase + r + g8;
        dv[r] = (e < E) ? dst[e] : -1;
    }
#pragma unroll
    for (int t = 0; t < 8; ++t) {
        float bv = b2[t * 16 + col];
#pragma unroll
        for (int r = 0; r < 8; ++r) {
            if (dv[r] >= 0)
                atomicAdd(accum + ((long)dv[r] * 128 + t * 16 + col),
                          acc2[t][r] + bv);
        }
    }
}

// ---- node ReLU + bf16 re-encode (+ fused graph pooling on last layer) -----

__global__ void node_update(const float* __restrict__ accum,
                            unsigned short* __restrict__ nodef, long n128,
                            const int* __restrict__ batch,
                            float* __restrict__ pool, float* __restrict__ cnt,
                            int do_pool) {
    long idx = blockIdx.x * (long)blockDim.x + threadIdx.x;
    if (idx >= n128) return;
    float v = accum[idx];
    v = v > 0.f ? v : 0.f;
    nodef[idx] = f2bf(v);
    if (do_pool) {
        long nn = idx >> 7;
        int  h  = (int)(idx & 127);
        int  g  = batch[nn];
        atomicAdd(pool + (long)g * 128 + h, v);
        if (h == 0) atomicAdd(cnt + g, 1.0f);
    }
}

__global__ void final_linear(const float* __restrict__ pool,
                             const float* __restrict__ cnt,
                             const float* __restrict__ lw,
                             const float* __restrict__ lb,
                             float* __restrict__ out) {
    int g = blockIdx.x;
    int o = threadIdx.x;
    if (o >= 8) return;
    float c = cnt[g];
    if (c < 1.f) c = 1.f;
    const float* pg = pool + (long)g * 128;
    float s = 0.f;
    for (int h = 0; h < 128; ++h) s += pg[h] * lw[h * 8 + o];
    out[g * 8 + o] = s / c + lb[o];
}

// ------------------------------- launcher ----------------------------------

extern "C" void kernel_launch(void* const* d_in, const int* in_sizes, int n_in,
                              void* d_out, int out_size, void* d_ws,
                              size_t ws_size, hipStream_t stream) {
    const float* x     = (const float*)d_in[0];
    const int*   ei    = (const int*)d_in[1];
    const float* ea    = (const float*)d_in[2];
    const int*   batch = (const int*)d_in[3];
    const float* W[3][4];
    for (int l = 0; l < 3; ++l)
        for (int j = 0; j < 4; ++j) W[l][j] = (const float*)d_in[4 + l * 4 + j];
    const float* lw  = (const float*)d_in[16];
    const float* lb  = (const float*)d_in[17];
    float*       out = (float*)d_out;

    const long N = in_sizes[0] / 64;
    const long E = in_sizes[1] / 2;
    const int* src = ei;
    const int* dst = ei + E;

    char*  ws  = (char*)d_ws;
    size_t off = 0;
    auto alloc = [&](size_t bytes) {
        size_t o = off;
        off += (bytes + 255) & ~(size_t)255;
        return o;
    };
    unsigned short* nodef = (unsigned short*)(ws + alloc((size_t)N * 128 * 2));
    float*          accum = (float*)(ws + alloc((size_t)N * 128 * 4));
    unsigned short* ef    = (unsigned short*)(ws + alloc((size_t)E * 32 * 2));
    const int nkb1[3] = {3, 5, 5};   // (96, 160, 160) / 32
    unsigned short* wp[3][2];
    for (int l = 0; l < 3; ++l) {
        wp[l][0] = (unsigned short*)(ws + alloc((size_t)nkb1[l] * 4096 * 2));
        wp[l][1] = (unsigned short*)(ws + alloc((size_t)4 * 4096 * 2));
    }
    float* pool = (float*)(ws + alloc(128 * 128 * 4));
    float* cnt  = (float*)(ws + alloc(128 * 4));

    auto gz = [](long n) { return (unsigned)((n + 255) / 256); };

    // one-time (per call) conversions / packing
    cvt_f32_bf16<<<gz(N * 64), 256, 0, stream>>>(x, nodef, N * 64);
    cvt_f32_bf16<<<gz(E * 32), 256, 0, stream>>>(ea, ef, E * 32);
    for (int l = 0; l < 3; ++l) {
        pack_w_bf16<<<gz((long)nkb1[l] * 4096), 256, 0, stream>>>(W[l][0],
                                                                  wp[l][0],
                                                                  nkb1[l]);
        pack_w_bf16<<<gz(4L * 4096), 256, 0, stream>>>(W[l][2], wp[l][1], 4);
    }
    zero_f32<<<gz(128 * 128), 256, 0, stream>>>(pool, 128 * 128);
    zero_f32<<<1, 256, 0, stream>>>(cnt, 128);

    const unsigned eblocks = (unsigned)(((E + 15) / 16 + 3) / 4);

    for (int l = 0; l < 3; ++l) {
        zero_f32<<<gz(N * 128), 256, 0, stream>>>(accum, N * 128);
        if (l == 0)
            edge_mlp_kernel<64><<<eblocks, 128, 0, stream>>>(
                nodef, ef, src, dst, wp[0][0], W[0][1], wp[0][1], W[0][3],
                accum, (int)E);
        else
            edge_mlp_kernel<128><<<eblocks, 128, 0, stream>>>(
                nodef, ef, src, dst, wp[l][0], W[l][1], wp[l][1], W[l][3],
                accum, (int)E);
        node_update<<<gz(N * 128), 256, 0, stream>>>(
            accum, nodef, N * 128, batch, pool, cnt, (l == 2) ? 1 : 0);
    }
    final_linear<<<128, 32, 0, stream>>>(pool, cnt, lw, lb, out);
}
